// VARTransformer_4294967296507
// MI455X (gfx1250) — matrix operand order repaired and checked by reading, not compiled, not run on hardware
//
#include <hip/hip_runtime.h>
#include <hip/hip_bf16.h>

// ---------------- problem constants ----------------
#define NLAY 12
#define D_   1024
#define H_   16
#define DHD  64
#define DFF_ 2752
#define NS_  6
#define BB_  2
#define TT_  1365
#define BT_  (BB_*TT_)
#define VV_  4096

typedef __attribute__((ext_vector_type(16))) __bf16 v16bf;
typedef __attribute__((ext_vector_type(4)))  __bf16 v4bf;
typedef __attribute__((ext_vector_type(8)))  float  v8f;

__constant__ int d_start[NS_+1] = {0, 1, 5, 21, 85, 341, 1365};

__device__ __forceinline__ int scale_of(int t) {
  return (t >= 341) ? 5 : (t >= 85) ? 4 : (t >= 21) ? 3 : (t >= 5) ? 2 : (t >= 1) ? 1 : 0;
}

__device__ __forceinline__ v8f wmma_bf16(v16bf a, v16bf b, v8f c) {
  return __builtin_amdgcn_wmma_f32_16x16x32_bf16(false, a, false, b, (short)0, c, false, false);
}

// Permuted k-position so the ISA A-fragment (k = hf*8 + e%8 + (e<8?0:16)) reads
// 16 *contiguous* bf16 per lane:  pos(k) = ((k>>3)&1)*16 + (k>>4)*8 + (k&7)
__device__ __forceinline__ int aperm(int k) {
  return (((k >> 3) & 1) << 4) + ((k >> 4) << 3) + (k & 7);
}

__device__ __forceinline__ v4bf pack4(float x, float y, float z, float w) {
  v4bf r; r[0] = (__bf16)x; r[1] = (__bf16)y; r[2] = (__bf16)z; r[3] = (__bf16)w;
  return r;
}

// 16 contiguous floats -> v16bf
__device__ __forceinline__ v16bf cvt16(const float* __restrict__ p) {
  float4 a = *(const float4*)(p);
  float4 b = *(const float4*)(p + 4);
  float4 c = *(const float4*)(p + 8);
  float4 d = *(const float4*)(p + 12);
  v16bf r;
  r[0]=(__bf16)a.x;  r[1]=(__bf16)a.y;  r[2]=(__bf16)a.z;  r[3]=(__bf16)a.w;
  r[4]=(__bf16)b.x;  r[5]=(__bf16)b.y;  r[6]=(__bf16)b.z;  r[7]=(__bf16)b.w;
  r[8]=(__bf16)c.x;  r[9]=(__bf16)c.y;  r[10]=(__bf16)c.z; r[11]=(__bf16)c.w;
  r[12]=(__bf16)d.x; r[13]=(__bf16)d.y; r[14]=(__bf16)d.z; r[15]=(__bf16)d.w;
  return r;
}

// two 8-float chunks (p0: elems 0..7, p1: elems 8..15), scaled -> v16bf
__device__ __forceinline__ v16bf cvt8x2s(const float* __restrict__ p0,
                                         const float* __restrict__ p1, float s) {
  float4 a = *(const float4*)(p0);
  float4 b = *(const float4*)(p0 + 4);
  float4 c = *(const float4*)(p1);
  float4 d = *(const float4*)(p1 + 4);
  v16bf r;
  r[0]=(__bf16)(a.x*s);  r[1]=(__bf16)(a.y*s);  r[2]=(__bf16)(a.z*s);  r[3]=(__bf16)(a.w*s);
  r[4]=(__bf16)(b.x*s);  r[5]=(__bf16)(b.y*s);  r[6]=(__bf16)(b.z*s);  r[7]=(__bf16)(b.w*s);
  r[8]=(__bf16)(c.x*s);  r[9]=(__bf16)(c.y*s);  r[10]=(__bf16)(c.z*s); r[11]=(__bf16)(c.w*s);
  r[12]=(__bf16)(d.x*s); r[13]=(__bf16)(d.y*s); r[14]=(__bf16)(d.z*s); r[15]=(__bf16)(d.w*s);
  return r;
}

struct IdxPtrs { const int* p[6]; };

// ---------------- GEMM: C[M,N] = A[M,K] x B ----------------
// TRB=false: B is [K,N] row-major (ldb = N).  TRB=true: B is [N,K] row-major (ldb = K).
// Requires N % 64 == 0, K % 32 == 0. M guarded.
#define BMT 64
#define BNT 64
#define BKT 32
#define LSTR 48   // LDS row stride (elements): rows 96B -> all 32B chunks aligned

template<bool TRB>
__global__ void __launch_bounds__(128)
gemm_kernel(const float* __restrict__ A, int lda,
            const float* __restrict__ Bm, int ldb,
            float* __restrict__ C, int ldc,
            int M, int N, int K) {
  __shared__ __bf16 As[BMT * LSTR];   // As[m][k] plain layout
  __shared__ __bf16 Bs[BNT * LSTR];   // Bs[n][k] plain layout
  const int tid  = threadIdx.x;
  const int wave = tid >> 5;
  const int lane = tid & 31;
  const int hf   = lane >> 4;
  const int ln16 = lane & 15;
  const int row0 = blockIdx.y * BMT;
  const int col0 = blockIdx.x * BNT;
  const int wr   = (wave >> 1) * 32;   // wave row offset within tile
  const int wc   = (wave & 1) * 32;    // wave col offset within tile

  v8f acc[2][2];
  #pragma unroll
  for (int i = 0; i < 2; i++)
    #pragma unroll
    for (int j = 0; j < 2; j++)
      #pragma unroll
      for (int r = 0; r < 8; r++) acc[i][j][r] = 0.f;

  for (int k0 = 0; k0 < K; k0 += BKT) {
    // --- stage A tile (BMT x BKT) fp32->bf16, plain k, v4bf stores ---
    {
      const int rr = tid >> 3;                 // 0..15
      const int c4 = (tid & 7) * 4;            // 0,4,...,28
      const int dp = c4;                       // plain k position
      #pragma unroll
      for (int p = 0; p < 4; p++) {
        const int r  = rr + p * 16;
        const int gr = row0 + r;
        float4 val = make_float4(0.f, 0.f, 0.f, 0.f);
        if (gr < M) val = *(const float4*)(A + (size_t)gr * lda + k0 + c4);
        *(v4bf*)(&As[r * LSTR + dp]) = pack4(val.x, val.y, val.z, val.w);
      }
    }
    // --- stage B tile into Bs[n][k] ---
    if (TRB) {
      const int nn = tid >> 3;
      const int c4 = (tid & 7) * 4;
      #pragma unroll
      for (int p = 0; p < 4; p++) {
        const int n = nn + p * 16;
        float4 val = *(const float4*)(Bm + (size_t)(col0 + n) * ldb + k0 + c4);
        *(v4bf*)(&Bs[n * LSTR + c4]) = pack4(val.x, val.y, val.z, val.w);
      }
    } else {
      // 4x4 block transpose: 4 float4 loads + 4 v4bf stores per thread
      const int kb = (tid >> 4) * 4;           // 0..28
      const int nb = (tid & 15) * 4;           // 0..60
      float4 r0 = *(const float4*)(Bm + (size_t)(k0 + kb + 0) * ldb + col0 + nb);
      float4 r1 = *(const float4*)(Bm + (size_t)(k0 + kb + 1) * ldb + col0 + nb);
      float4 r2 = *(const float4*)(Bm + (size_t)(k0 + kb + 2) * ldb + col0 + nb);
      float4 r3 = *(const float4*)(Bm + (size_t)(k0 + kb + 3) * ldb + col0 + nb);
      *(v4bf*)(&Bs[(nb + 0) * LSTR + kb]) = pack4(r0.x, r1.x, r2.x, r3.x);
      *(v4bf*)(&Bs[(nb + 1) * LSTR + kb]) = pack4(r0.y, r1.y, r2.y, r3.y);
      *(v4bf*)(&Bs[(nb + 2) * LSTR + kb]) = pack4(r0.z, r1.z, r2.z, r3.z);
      *(v4bf*)(&Bs[(nb + 3) * LSTR + kb]) = pack4(r0.w, r1.w, r2.w, r3.w);
    }
    __syncthreads();

    // --- fragments: single contiguous 32B LDS loads ---
    v16bf af[2], bfr[2];
    #pragma unroll
    for (int i = 0; i < 2; i++)
      af[i] = *(const v16bf*)(&As[(wr + i * 16 + ln16) * LSTR + hf * 16]);
    #pragma unroll
    for (int j = 0; j < 2; j++)
      bfr[j] = *(const v16bf*)(&Bs[(wc + j * 16 + ln16) * LSTR + hf * 16]);
    #pragma unroll
    for (int i = 0; i < 2; i++)
      #pragma unroll
      for (int j = 0; j < 2; j++)
        acc[i][j] = wmma_bf16(af[i], bfr[j], acc[i][j]);
    __syncthreads();
  }

  // --- epilogue: lane -> n = ln16, elem r -> m = r + 8*hf ---
  #pragma unroll
  for (int i = 0; i < 2; i++)
    #pragma unroll
    for (int j = 0; j < 2; j++)
      #pragma unroll
      for (int r = 0; r < 8; r++) {
        const int grow = row0 + wr + i * 16 + r + 8 * hf;
        const int gcol = col0 + wc + j * 16 + ln16;
        if (grow < M) C[(size_t)grow * ldc + gcol] = acc[i][j][r];
      }
}

// ---------------- embedding ----------------
__global__ void embed_kernel(IdxPtrs idx, const float* __restrict__ te,
                             const float* __restrict__ se,
                             const float* __restrict__ st,
                             float* __restrict__ hbuf) {
  const int i = blockIdx.x * blockDim.x + threadIdx.x;
  if (i >= BT_ * D_) return;
  const int row = i / D_, d = i % D_;
  const int b = row / TT_, t = row % TT_;
  const int s = scale_of(t);
  float base;
  if (t == 0) base = st[d];
  else {
    const int sz  = d_start[s + 1] - d_start[s];
    const int tok = idx.p[s][b * sz + (t - d_start[s])];
    base = te[(size_t)tok * D_ + d];
  }
  hbuf[i] = base + se[s * D_ + d];
}

// ---------------- per-head QK RMSNorm + RoPE ----------------
__global__ void __launch_bounds__(64)
qkrope_kernel(float* __restrict__ q, float* __restrict__ k,
              const float* __restrict__ qg, const float* __restrict__ kg) {
  __shared__ float red[64];
  __shared__ float xs[64];
  const int d  = threadIdx.x;
  const int bt = blockIdx.x;
  const int h  = blockIdx.y;
  const int t  = bt % TT_;
  const int s  = scale_of(t);
  const float pos = (float)(t - d_start[s]);
  const int   j   = d & 31;
  const float ang = pos * __powf(10000.f, -(float)(2 * j) / (float)DHD);
  const float cs = __cosf(ang), sn = __sinf(ang);
  const size_t base = (size_t)bt * D_ + h * DHD;

  // Q
  float xv = q[base + d];
  red[d] = xv * xv; __syncthreads();
  for (int off = 32; off > 0; off >>= 1) { if (d < off) red[d] += red[d + off]; __syncthreads(); }
  float nq = xv * rsqrtf(red[0] / (float)DHD + 1e-6f) * qg[d];
  __syncthreads();
  xs[d] = nq; __syncthreads();
  float rot = (d < 32) ? -xs[d + 32] : xs[d - 32];
  q[base + d] = nq * cs + rot * sn;
  __syncthreads();

  // K
  xv = k[base + d];
  red[d] = xv * xv; __syncthreads();
  for (int off = 32; off > 0; off >>= 1) { if (d < off) red[d] += red[d + off]; __syncthreads(); }
  float nk = xv * rsqrtf(red[0] / (float)DHD + 1e-6f) * kg[d];
  __syncthreads();
  xs[d] = nk; __syncthreads();
  rot = (d < 32) ? -xs[d + 32] : xs[d - 32];
  k[base + d] = nk * cs + rot * sn;
}

// ---------------- flash attention (1 wave / 16-row tile / (b,h)) ----------------
__global__ void __launch_bounds__(32)
attn_kernel(const float* __restrict__ q, const float* __restrict__ k,
            const float* __restrict__ v, float* __restrict__ o) {
  __shared__ __bf16 Pl[16 * LSTR];   // Pl[m][kk] plain layout
  __shared__ __bf16 Vs[64 * LSTR];   // Vs[d][kk] plain layout
  const int lane = threadIdx.x;
  const int hf   = lane >> 4;
  const int ln16 = lane & 15;
  const int i0 = blockIdx.x * 16;
  const int h  = blockIdx.y;
  const int b  = blockIdx.z;

  int rlim[8];
  #pragma unroll
  for (int r = 0; r < 8; r++) {
    int row = i0 + r + 8 * hf;
    if (row > TT_ - 1) row = TT_ - 1;
    rlim[r] = d_start[scale_of(row) + 1];     // block-causal mask == per-row column limit
  }
  int limMax;
  { int rl = (i0 + 15 < TT_) ? i0 + 15 : TT_ - 1; limMax = d_start[scale_of(rl) + 1]; }
  const int npairs = (limMax + 31) >> 5;

  // Q fragments, pre-scaled by 1/sqrt(DH)=0.125
  int iq = i0 + ln16; if (iq > TT_ - 1) iq = TT_ - 1;
  const float* qrow = q + ((size_t)(b * TT_ + iq) * D_ + h * DHD);
  const v16bf qf0 = cvt8x2s(qrow + hf * 16,      qrow + hf * 16 + 8,      0.125f);
  const v16bf qf1 = cvt8x2s(qrow + 32 + hf * 16, qrow + 40 + hf * 16,      0.125f);

  float mrun[8], lrun[8];
  #pragma unroll
  for (int r = 0; r < 8; r++) { mrun[r] = -3.0e38f; lrun[r] = 0.f; }
  v8f oacc[4];
  #pragma unroll
  for (int nt = 0; nt < 4; nt++)
    #pragma unroll
    for (int r = 0; r < 8; r++) oacc[nt][r] = 0.f;

  // P-store positions (plain k) for this lane's two subtile columns
  const int pp0 = ln16;               // sub 0: kloc = ln16
  const int pp1 = 16 + ln16;          // sub 1: kloc = 16 + ln16

  for (int pt = 0; pt < npairs; pt++) {
    const int c0 = pt * 32;
    __syncthreads();
    // stage V tile (32 keys x 64 dims) transposed via 4x4 blocks: Vs[d][kk]
    #pragma unroll
    for (int p = 0; p < 4; p++) {
      const int bi = p * 32 + lane;            // 0..127 blocks
      const int kb = (bi >> 4) * 4;            // 0..28
      const int db = (bi & 15) * 4;            // 0..60
      int s0i = c0 + kb;
      int s0c = (s0i + 0 > TT_ - 1) ? TT_ - 1 : s0i + 0;
      int s1c = (s0i + 1 > TT_ - 1) ? TT_ - 1 : s0i + 1;
      int s2c = (s0i + 2 > TT_ - 1) ? TT_ - 1 : s0i + 2;
      int s3c = (s0i + 3 > TT_ - 1) ? TT_ - 1 : s0i + 3;
      float4 r0 = *(const float4*)(v + ((size_t)(b * TT_ + s0c) * D_ + h * DHD + db));
      float4 r1 = *(const float4*)(v + ((size_t)(b * TT_ + s1c) * D_ + h * DHD + db));
      float4 r2 = *(const float4*)(v + ((size_t)(b * TT_ + s2c) * D_ + h * DHD + db));
      float4 r3 = *(const float4*)(v + ((size_t)(b * TT_ + s3c) * D_ + h * DHD + db));
      *(v4bf*)(&Vs[(db + 0) * LSTR + kb]) = pack4(r0.x, r1.x, r2.x, r3.x);
      *(v4bf*)(&Vs[(db + 1) * LSTR + kb]) = pack4(r0.y, r1.y, r2.y, r3.y);
      *(v4bf*)(&Vs[(db + 2) * LSTR + kb]) = pack4(r0.z, r1.z, r2.z, r3.z);
      *(v4bf*)(&Vs[(db + 3) * LSTR + kb]) = pack4(r0.w, r1.w, r2.w, r3.w);
    }

    // S = Q K^T for two 16-column subtiles
    float s0[8], s1[8];
    #pragma unroll
    for (int sub = 0; sub < 2; sub++) {
      const int c = c0 + sub * 16;
      int colc = c + ln16; if (colc > TT_ - 1) colc = TT_ - 1;
      const float* krow = k + ((size_t)(b * TT_ + colc) * D_ + h * DHD);
      const v16bf kf0 = cvt16(krow + hf * 16);
      const v16bf kf1 = cvt16(krow + 32 + hf * 16);
      v8f sacc;
      #pragma unroll
      for (int r = 0; r < 8; r++) sacc[r] = 0.f;
      sacc = wmma_bf16(qf0, kf0, sacc);
      sacc = wmma_bf16(qf1, kf1, sacc);
      const int col = c + ln16;
      #pragma unroll
      for (int r = 0; r < 8; r++) {
        const float sv = (col < rlim[r]) ? sacc[r] : -3.0e38f;
        if (sub == 0) s0[r] = sv; else s1[r] = sv;
      }
    }

    // online softmax across both subtiles; P -> LDS (plain-k layout)
    #pragma unroll
    for (int r = 0; r < 8; r++) {
      float tmax = fmaxf(s0[r], s1[r]);
      #pragma unroll
      for (int off = 1; off < 16; off <<= 1) tmax = fmaxf(tmax, __shfl_xor(tmax, off, 32));
      const float mnew = fmaxf(mrun[r], tmax);
      const float corr = __expf(mrun[r] - mnew);
      const float p0 = __expf(s0[r] - mnew);
      const float p1 = __expf(s1[r] - mnew);
      float psum = p0 + p1;
      #pragma unroll
      for (int off = 1; off < 16; off <<= 1) psum += __shfl_xor(psum, off, 32);
      mrun[r] = mnew;
      lrun[r] = lrun[r] * corr + psum;
      #pragma unroll
      for (int nt = 0; nt < 4; nt++) oacc[nt][r] = oacc[nt][r] * corr;
      const int m = r + 8 * hf;
      Pl[m * LSTR + pp0] = (__bf16)p0;
      Pl[m * LSTR + pp1] = (__bf16)p1;
    }
    __syncthreads();

    // O += P V  (single 32B fragment loads)
    const v16bf pf = *(const v16bf*)(&Pl[ln16 * LSTR + hf * 16]);
    #pragma unroll
    for (int nt = 0; nt < 4; nt++) {
      const v16bf vf = *(const v16bf*)(&Vs[(nt * 16 + ln16) * LSTR + hf * 16]);
      oacc[nt] = wmma_bf16(pf, vf, oacc[nt]);
    }
  }

  // write O / l
  #pragma unroll
  for (int nt = 0; nt < 4; nt++)
    #pragma unroll
    for (int r = 0; r < 8; r++) {
      const int row = i0 + r + 8 * hf;
      if (row < TT_) {
        o[(size_t)(b * TT_ + row) * D_ + h * DHD + nt * 16 + ln16] = oacc[nt][r] / lrun[r];
      }
    }
}

// ---------------- residual + RMSNorm (post-norm) ----------------
__global__ void __launch_bounds__(256)
add_rms_kernel(float* __restrict__ hbuf, const float* __restrict__ delta,
               const float* __restrict__ gain) {
  __shared__ float red[256];
  const int row = blockIdx.x;
  const int tid = threadIdx.x;
  float x[4]; float ss = 0.f;
  #pragma unroll
  for (int i = 0; i < 4; i++) {
    const int d = tid + i * 256;
    const float vv = hbuf[(size_t)row * D_ + d] + delta[(size_t)row * D_ + d];
    x[i] = vv; ss += vv * vv;
  }
  red[tid] = ss; __syncthreads();
  for (int off = 128; off > 0; off >>= 1) { if (tid < off) red[tid] += red[tid + off]; __syncthreads(); }
  const float sc = rsqrtf(red[0] / (float)D_ + 1e-6f);
  #pragma unroll
  for (int i = 0; i < 4; i++) {
    const int d = tid + i * 256;
    hbuf[(size_t)row * D_ + d] = x[i] * sc * gain[d];
  }
}

// ---------------- SwiGLU gate ----------------
__global__ void silu_mul_kernel(float* __restrict__ g, const float* __restrict__ u, int n) {
  const int i = blockIdx.x * blockDim.x + threadIdx.x;
  if (i < n) { const float x = g[i]; g[i] = (x / (1.f + __expf(-x))) * u[i]; }
}

// ---------------- loss ----------------
__global__ void zero_kernel(float* acc) { if (threadIdx.x < 2) acc[threadIdx.x] = 0.f; }

__global__ void __launch_bounds__(256)
loss_row_kernel(const float* __restrict__ logits, IdxPtrs idx, float* __restrict__ acc) {
  __shared__ float red[256];
  const int row = blockIdx.x, tid = threadIdx.x;
  const float* lr = logits + (size_t)row * VV_;
  float mx = -3.0e38f;
  for (int i = tid; i < VV_; i += 256) mx = fmaxf(mx, lr[i]);
  red[tid] = mx; __syncthreads();
  for (int off = 128; off > 0; off >>= 1) { if (tid < off) red[tid] = fmaxf(red[tid], red[tid + off]); __syncthreads(); }
  mx = red[0]; __syncthreads();
  float se = 0.f;
  for (int i = tid; i < VV_; i += 256) se += __expf(lr[i] - mx);
  red[tid] = se; __syncthreads();
  for (int off = 128; off > 0; off >>= 1) { if (tid < off) red[tid] += red[tid + off]; __syncthreads(); }
  if (tid == 0) {
    const float lse = mx + __logf(red[0]);
    const int b = row / TT_, t = row % TT_;
    const int s = scale_of(t);
    const int sz = d_start[s + 1] - d_start[s];
    const int tok = idx.p[s][b * sz + (t - d_start[s])];
    atomicAdd(&acc[0], -(lr[tok] - lse));
    atomicAdd(&acc[1], lse * lse);
  }
}

__global__ void finalize_kernel(const float* __restrict__ acc, float* __restrict__ out) {
  if (threadIdx.x == 0 && blockIdx.x == 0)
    out[0] = acc[0] / (float)BT_ + 1e-4f * (acc[1] / (float)BT_);
}

// ---------------- host launcher ----------------
static void launch_gemm(bool btrans, const float* A, int lda, const float* Bm, int ldb,
                        float* C, int ldc, int M, int N, int K, hipStream_t s) {
  dim3 grid(N / BNT, (M + BMT - 1) / BMT);
  if (btrans) gemm_kernel<true ><<<grid, 128, 0, s>>>(A, lda, Bm, ldb, C, ldc, M, N, K);
  else        gemm_kernel<false><<<grid, 128, 0, s>>>(A, lda, Bm, ldb, C, ldc, M, N, K);
}

extern "C" void kernel_launch(void* const* d_in, const int* in_sizes, int n_in,
                              void* d_out, int out_size, void* d_ws, size_t ws_size,
                              hipStream_t stream) {
  IdxPtrs idx;
  for (int i = 0; i < 6; i++) idx.p[i] = (const int*)d_in[i];
  const float* te = (const float*)d_in[6];
  const float* se = (const float*)d_in[7];
  const float* st = (const float*)d_in[8];
  const float* wq = (const float*)d_in[9];
  const float* wk = (const float*)d_in[10];
  const float* wv = (const float*)d_in[11];
  const float* wo = (const float*)d_in[12];
  const float* qn = (const float*)d_in[13];
  const float* kn = (const float*)d_in[14];
  const float* n1 = (const float*)d_in[15];
  const float* n2 = (const float*)d_in[16];
  const float* w1 = (const float*)d_in[17];
  const float* w3 = (const float*)d_in[18];
  const float* w2 = (const float*)d_in[19];

  float* ws = (float*)d_ws;
  const size_t nHD = (size_t)BT_ * D_;
  const size_t nF  = (size_t)BT_ * DFF_;
  const size_t nLG = (size_t)BT_ * VV_;
  float* hb  = ws;
  float* qb  = hb + nHD;
  float* kb  = qb + nHD;
  float* vb  = kb + nHD;
  float* tb  = vb + nHD;
  float* gb  = tb + nHD;
  float* ub  = gb + nF;
  float* lg  = ub + nF;
  float* acc = lg + nLG;

  embed_kernel<<<(BT_ * D_ + 255) / 256, 256, 0, stream>>>(idx, te, se, st, hb);

  for (int l = 0; l < NLAY; l++) {
    const float* wql = wq + (size_t)l * D_ * D_;
    const float* wkl = wk + (size_t)l * D_ * D_;
    const float* wvl = wv + (size_t)l * D_ * D_;
    const float* wol = wo + (size_t)l * D_ * D_;
    const float* w1l = w1 + (size_t)l * D_ * DFF_;
    const float* w3l = w3 + (size_t)l * D_ * DFF_;
    const float* w2l = w2 + (size_t)l * DFF_ * D_;

    launch_gemm(false, hb, D_, wql, D_, qb, D_, BT_, D_, D_, stream);
    launch_gemm(false, hb, D_, wkl, D_, kb, D_, BT_, D_, D_, stream);
    launch_gemm(false, hb, D_, wvl, D_, vb, D_, BT_, D_, D_, stream);

    qkrope_kernel<<<dim3(BT_, H_), 64, 0, stream>>>(qb, kb, qn + l * DHD, kn + l * DHD);

    attn_kernel<<<dim3((TT_ + 15) / 16, H_, BB_), 32, 0, stream>>>(qb, kb, vb, tb);

    launch_gemm(false, tb, D_, wol, D_, qb, D_, BT_, D_, D_, stream);   // qb = attn-out proj
    add_rms_kernel<<<BT_, 256, 0, stream>>>(hb, qb, n1 + l * D_);

    launch_gemm(false, hb, D_, w1l, DFF_, gb, DFF_, BT_, DFF_, D_, stream);
    launch_gemm(false, hb, D_, w3l, DFF_, ub, DFF_, BT_, DFF_, D_, stream);
    silu_mul_kernel<<<(BT_ * DFF_ + 255) / 256, 256, 0, stream>>>(gb, ub, BT_ * DFF_);
    launch_gemm(false, gb, DFF_, w2l, D_, tb, D_, BT_, D_, DFF_, stream);
    add_rms_kernel<<<BT_, 256, 0, stream>>>(hb, tb, n2 + l * D_);
  }

  // tied head: logits = h @ token_embed^T  (B is [V, D] row-major -> TRB)
  launch_gemm(true, hb, D_, te, D_, lg, VV_, BT_, VV_, D_, stream);

  zero_kernel<<<1, 32, 0, stream>>>(acc);
  loss_row_kernel<<<BT_, 256, 0, stream>>>(lg, idx, acc);
  finalize_kernel<<<1, 32, 0, stream>>>(acc, (float*)d_out);
}
